// User_State_Intensity_Model_50611894616436
// MI455X (gfx1250) — matrix operand order repaired and checked by reading, not compiled, not run on hardware
//
#include <hip/hip_runtime.h>
#include <hip/hip_bf16.h>
#include <math.h>

// ---------------------------------------------------------------------------
// Fused User_State_Intensity_Model integral for MI455X (gfx1250, wave32).
//   131 steps of:  s *= exp(-0.01);  acc += relu(silu(silu(s@W1+b1)@W2+b2)@W3+b3)*dt
// GEMMs run on v_wmma_f32_16x16x32_bf16. State stays f32 in LDS the whole time;
// the decay multiply is fused into the GEMM1 A-fragment load (each state element
// is touched exactly once per step).
// ---------------------------------------------------------------------------

typedef __attribute__((ext_vector_type(16))) __bf16 v16bf;
typedef __attribute__((ext_vector_type(8)))  float  v8f;

#define S_DIM   128
#define H_DIM   256
#define ROWS    128      // rows of B per block
#define THREADS 256      // 8 waves of 32

// LDS layout (dynamic shared):
//   [0      , 65536)  float  sF32[128][128]   state tile, f32 master copy
//   [65536  , 131072) __bf16 h1  [128][256]   GEMM1 output (per-wave 16-row strips)
//   [131072 , 196608) __bf16 W1T [256][128]   W1 transposed, bf16
//   [196608 , ...)    float  b1s[256], b2s[256], w3s[256]
#define SMEM_BYTES (65536 + 65536 + 65536 + 3 * H_DIM * 4)

static __device__ __forceinline__ unsigned pk_bf16(float lo, float hi) {
  // pack two f32 -> two bf16 (truncate) in one dword
  return (__float_as_uint(hi) & 0xFFFF0000u) | (__float_as_uint(lo) >> 16);
}

static __device__ __forceinline__ v16bf pack_afrag_f32(float4 a0, float4 a1,
                                                       float4 a2, float4 a3) {
  union { unsigned u[8]; v16bf v; } r;
  r.u[0] = pk_bf16(a0.x, a0.y); r.u[1] = pk_bf16(a0.z, a0.w);
  r.u[2] = pk_bf16(a1.x, a1.y); r.u[3] = pk_bf16(a1.z, a1.w);
  r.u[4] = pk_bf16(a2.x, a2.y); r.u[5] = pk_bf16(a2.z, a2.w);
  r.u[6] = pk_bf16(a3.x, a3.y); r.u[7] = pk_bf16(a3.z, a3.w);
  return r.v;
}

static __device__ __forceinline__ v16bf ld_frag(const __bf16* p) {
  union { uint4 u[2]; v16bf v; } r;
  r.u[0] = *(const uint4*)(p);
  r.u[1] = *(const uint4*)(p + 16);
  return r.v;
}

static __device__ __forceinline__ float fast_silu(float x) {
  return x * (1.0f / (1.0f + __expf(-x)));
}

// --- prep: transpose + convert weights to bf16 [N][K] so every WMMA B-frag
//     is two 16B reads from one row per lane ------------------------------------
__global__ void prep_weights_kernel(const float* __restrict__ W1,
                                    const float* __restrict__ W2,
                                    __bf16* __restrict__ W1T,
                                    __bf16* __restrict__ W2T) {
  int idx    = blockIdx.x * blockDim.x + threadIdx.x;
  int stride = gridDim.x * blockDim.x;
  for (int i = idx; i < H_DIM * S_DIM; i += stride) {    // W1T[n][k] = W1[k][n]
    int n = i / S_DIM, k = i - n * S_DIM;
    W1T[i] = (__bf16)W1[k * H_DIM + n];
  }
  for (int i = idx; i < H_DIM * H_DIM; i += stride) {    // W2T[n][k] = W2[k][n]
    int n = i / H_DIM, k = i - n * H_DIM;
    W2T[i] = (__bf16)W2[k * H_DIM + n];
  }
}

// --- main fused kernel ------------------------------------------------------
__global__ __launch_bounds__(THREADS)
void usim_fused_kernel(const int*   __restrict__ timep,
                       const float* __restrict__ state,
                       const __bf16* __restrict__ W1T_g,
                       const __bf16* __restrict__ W2T_g,
                       const float* __restrict__ b1,
                       const float* __restrict__ b2,
                       const float* __restrict__ W3,
                       const float* __restrict__ b3,
                       float* __restrict__ out) {
  extern __shared__ char smem[];
  float*  sF32 = (float*)(smem);
  __bf16* h1   = (__bf16*)(smem + 65536);
  __bf16* W1T  = (__bf16*)(smem + 131072);
  float*  b1s  = (float*)(smem + 196608);
  float*  b2s  = (float*)(smem + 196608 + H_DIM * 4);
  float*  w3s  = (float*)(smem + 196608 + 2 * H_DIM * 4);

  const int tid     = threadIdx.x;
  const int lane    = tid & 31;
  const int wave    = tid >> 5;          // 0..7, owns rows [wave*16, wave*16+16)
  const int wrow    = wave * 16;
  const int rowBase = blockIdx.x * ROWS;
  const int col16   = lane & 15;         // N / M index within a 16-tile
  const int half    = lane >> 4;         // lane-half selects K sub-block
  const int halfk8  = half * 8;

  // ---- one-time init --------------------------------------------------------
  {   // W1T -> LDS (64KB, coalesced b128 copies)
    const uint4* src = (const uint4*)W1T_g;
    uint4*       dst = (uint4*)W1T;
    #pragma unroll 4
    for (int i = tid; i < (H_DIM * S_DIM * 2) / 16; i += THREADS) dst[i] = src[i];
  }
  for (int i = tid; i < H_DIM; i += THREADS) {
    b1s[i] = b1[i]; b2s[i] = b2[i]; w3s[i] = W3[i];
  }
  {   // state tile: wave-private 16x128 f32 strip
    const uint4* s_src = (const uint4*)(state + (size_t)(rowBase + wrow) * S_DIM);
    uint4*       s_dst = (uint4*)(sF32 + wrow * S_DIM);
    #pragma unroll 4
    for (int i = lane; i < 16 * S_DIM / 4; i += 32) s_dst[i] = s_src[i];
  }
  __syncthreads();   // only barrier: publish shared W1T/biases

  // ---- step schedule (mirror Python double semantics) -----------------------
  const double t     = (double)timep[0];
  const int    nfull = (int)floor(t / 0.1);
  const double restd = fmod(t, 0.1);
  const int    nsteps = nfull + ((restd > 0.01) ? 1 : 0);
  const float  decay_full = __expf(-0.1f * 0.1f);
  const float  decay_rest = __expf((float)(-0.1 * restd));
  const float  dt_rest    = (float)restd;

  float*        arow1 = sF32 + (wrow + col16) * S_DIM;   // GEMM1 A source row
  const __bf16* arow2 = h1   + (wrow + col16) * H_DIM;   // GEMM2 A source row
  const float   b3v   = b3[0];

  float accv[8];
  #pragma unroll
  for (int r = 0; r < 8; ++r) accv[r] = 0.0f;

  for (int it = 0; it < nsteps; ++it) {
    const float decay = (it < nfull) ? decay_full : decay_rest;
    const float dt    = (it < nfull) ? 0.1f       : dt_rest;

    // -- phase 1: decay state (fused) + build all GEMM1 A-fragments ----------
    // Across 32 lanes x 4 k-tiles, every element of the wave's 16x128 f32
    // strip is read exactly once: scale by decay, write back the f32 master,
    // and pack the bf16 A-fragment in registers.
    v16bf afrag1[4];
    #pragma unroll
    for (int kt = 0; kt < 4; ++kt) {
      const int ka = kt * 32 + halfk8;
      float4 a0 = *(const float4*)(arow1 + ka);
      float4 a1 = *(const float4*)(arow1 + ka + 4);
      float4 a2 = *(const float4*)(arow1 + ka + 16);
      float4 a3 = *(const float4*)(arow1 + ka + 20);
      a0.x *= decay; a0.y *= decay; a0.z *= decay; a0.w *= decay;
      a1.x *= decay; a1.y *= decay; a1.z *= decay; a1.w *= decay;
      a2.x *= decay; a2.y *= decay; a2.z *= decay; a2.w *= decay;
      a3.x *= decay; a3.y *= decay; a3.z *= decay; a3.w *= decay;
      *(float4*)(arow1 + ka)      = a0;
      *(float4*)(arow1 + ka + 4)  = a1;
      *(float4*)(arow1 + ka + 16) = a2;
      *(float4*)(arow1 + ka + 20) = a3;
      afrag1[kt] = pack_afrag_f32(a0, a1, a2, a3);
    }

    // -- phase 2: h1 = silu(s @ W1 + b1), 16x256 per wave, K=128 -------------
    #pragma unroll
    for (int nc = 0; nc < 4; ++nc) {          // 4 chunks of 4 N-tiles
      v8f acc0 = {0,0,0,0,0,0,0,0}, acc1 = acc0, acc2 = acc0, acc3 = acc0;
      #pragma unroll
      for (int kt = 0; kt < 4; ++kt) {        // K = 4 * 32
        const int kb = kt * 32 + half * 8;
        v16bf bf0 = ld_frag(W1T + (nc * 64 + 0  + col16) * S_DIM + kb);
        v16bf bf1 = ld_frag(W1T + (nc * 64 + 16 + col16) * S_DIM + kb);
        v16bf bf2 = ld_frag(W1T + (nc * 64 + 32 + col16) * S_DIM + kb);
        v16bf bf3 = ld_frag(W1T + (nc * 64 + 48 + col16) * S_DIM + kb);
        acc0 = __builtin_amdgcn_wmma_f32_16x16x32_bf16(false, afrag1[kt], false, bf0, (short)0, acc0, false, false);
        acc1 = __builtin_amdgcn_wmma_f32_16x16x32_bf16(false, afrag1[kt], false, bf1, (short)0, acc1, false, false);
        acc2 = __builtin_amdgcn_wmma_f32_16x16x32_bf16(false, afrag1[kt], false, bf2, (short)0, acc2, false, false);
        acc3 = __builtin_amdgcn_wmma_f32_16x16x32_bf16(false, afrag1[kt], false, bf3, (short)0, acc3, false, false);
      }
      // bias + SiLU + store bf16 to wave-private h1 strip (row-major)
      v8f accs[4] = {acc0, acc1, acc2, acc3};
      #pragma unroll
      for (int nt = 0; nt < 4; ++nt) {
        const int col = nc * 64 + nt * 16 + col16;
        const float bb = b1s[col];
        #pragma unroll
        for (int r = 0; r < 8; ++r) {
          float x = accs[nt][r] + bb;
          h1[(wrow + halfk8 + r) * H_DIM + col] = (__bf16)fast_silu(x);
        }
      }
    }

    // -- phase 3: load all GEMM2 A-fragments once (wave-private h1 strip) ----
    v16bf afrag2[8];
    #pragma unroll
    for (int kt = 0; kt < 8; ++kt) {
      const int ka = kt * 32 + halfk8;
      union { uint4 u[2]; v16bf v; } af;
      af.u[0] = *(const uint4*)(arow2 + ka);        // K chunk [ka, ka+8)
      af.u[1] = *(const uint4*)(arow2 + ka + 16);   // K chunk [ka+16, ka+24)
      afrag2[kt] = af.v;
    }

    // -- phase 4: h2 = silu(h1 @ W2 + b2); dot with W3 on the fly ------------
    float dotr[8];
    #pragma unroll
    for (int r = 0; r < 8; ++r) dotr[r] = 0.0f;

    #pragma unroll
    for (int nc = 0; nc < 4; ++nc) {
      v8f acc0 = {0,0,0,0,0,0,0,0}, acc1 = acc0, acc2 = acc0, acc3 = acc0;
      #pragma unroll
      for (int kt = 0; kt < 8; ++kt) {        // K = 8 * 32
        const int kb = kt * 32 + half * 8;
        v16bf bf0 = ld_frag(W2T_g + (size_t)(nc * 64 + 0  + col16) * H_DIM + kb);
        v16bf bf1 = ld_frag(W2T_g + (size_t)(nc * 64 + 16 + col16) * H_DIM + kb);
        v16bf bf2 = ld_frag(W2T_g + (size_t)(nc * 64 + 32 + col16) * H_DIM + kb);
        v16bf bf3 = ld_frag(W2T_g + (size_t)(nc * 64 + 48 + col16) * H_DIM + kb);
        acc0 = __builtin_amdgcn_wmma_f32_16x16x32_bf16(false, afrag2[kt], false, bf0, (short)0, acc0, false, false);
        acc1 = __builtin_amdgcn_wmma_f32_16x16x32_bf16(false, afrag2[kt], false, bf1, (short)0, acc1, false, false);
        acc2 = __builtin_amdgcn_wmma_f32_16x16x32_bf16(false, afrag2[kt], false, bf2, (short)0, acc2, false, false);
        acc3 = __builtin_amdgcn_wmma_f32_16x16x32_bf16(false, afrag2[kt], false, bf3, (short)0, acc3, false, false);
      }
      // bias + SiLU, then GEMM3 as a VALU dot against W3 (0.3% of FLOPs)
      v8f accs[4] = {acc0, acc1, acc2, acc3};
      #pragma unroll
      for (int nt = 0; nt < 4; ++nt) {
        const int col = nc * 64 + nt * 16 + col16;
        const float bb  = b2s[col];
        const float w3v = w3s[col];
        #pragma unroll
        for (int r = 0; r < 8; ++r) {
          float x = accs[nt][r] + bb;
          dotr[r] += fast_silu(x) * w3v;
        }
      }
    }

    // reduce partial dots over the 16 lanes sharing each row (C layout:
    // reg r, lane l -> row (l/16)*8 + r, col l%16). XOR masks 1..8 stay
    // within each 16-lane half on wave32.
    #pragma unroll
    for (int m = 1; m <= 8; m <<= 1) {
      #pragma unroll
      for (int r = 0; r < 8; ++r) dotr[r] += __shfl_xor(dotr[r], m, 32);
    }
    #pragma unroll
    for (int r = 0; r < 8; ++r) {
      float x = dotr[r] + b3v;
      accv[r] += (x > 0.0f ? x : 0.0f) * dt;   // relu(...) * dt
    }
  }

  // ---- epilogue: sigmoid, lanes 0 and 16 hold rows [0..8) / [8..16) --------
  if (col16 == 0) {
    const int rbase = rowBase + wrow + halfk8;
    #pragma unroll
    for (int r = 0; r < 8; ++r) {
      out[rbase + r] = 1.0f / (1.0f + __expf(-accv[r]));
    }
  }
}

// ---------------------------------------------------------------------------
extern "C" void kernel_launch(void* const* d_in, const int* in_sizes, int n_in,
                              void* d_out, int out_size, void* d_ws, size_t ws_size,
                              hipStream_t stream) {
  const int*   timep = (const int*)d_in[0];
  const float* state = (const float*)d_in[1];
  const float* W1    = (const float*)d_in[2];
  const float* b1    = (const float*)d_in[3];
  const float* W2    = (const float*)d_in[4];
  const float* b2    = (const float*)d_in[5];
  const float* W3    = (const float*)d_in[6];
  const float* b3    = (const float*)d_in[7];
  float*       out   = (float*)d_out;

  // scratch: bf16 transposed weights (64KB + 128KB)
  __bf16* W1T = (__bf16*)d_ws;
  __bf16* W2T = (__bf16*)((char*)d_ws + (size_t)H_DIM * S_DIM * sizeof(__bf16));

  prep_weights_kernel<<<128, 256, 0, stream>>>(W1, W2, W1T, W2T);

  const int Brows  = in_sizes[1] / S_DIM;     // 65536
  const int blocks = Brows / ROWS;            // 512
  usim_fused_kernel<<<blocks, THREADS, SMEM_BYTES, stream>>>(
      timep, state, W1T, W2T, b1, b2, W3, b3, out);
}
